// SparseSelfAttention_58042188038489
// MI455X (gfx1250) — matrix operand order repaired and checked by reading, not compiled, not run on hardware
//
#include <hip/hip_runtime.h>

typedef __attribute__((ext_vector_type(16))) _Float16 v16h;
typedef __attribute__((ext_vector_type(8)))  _Float16 v8h;
typedef __attribute__((ext_vector_type(4)))  _Float16 v4h;
typedef __attribute__((ext_vector_type(8)))  float    v8f;
typedef __attribute__((ext_vector_type(4)))  float    f4;
typedef __attribute__((ext_vector_type(4)))  unsigned int v4u;
typedef __attribute__((ext_vector_type(4)))  int      v4i;
typedef __attribute__((ext_vector_type(8)))  int      v8i;

#define BSZ  2
#define NH   8
#define SEQ  4096
#define DIM  64
#define QBLK 32
#define WIN  16
#define NB   (SEQ / QBLK)

// LDS row strides (in halves): 16B-aligned, bank-conflict-friendly.
#define KSTR 72   // Q/K f16 tiles: 32 rows x 64 halves (+8 pad)
#define VSTR 40   // V^T f16 tile : 64 rows x 32 halves (+8 pad)

#if __has_builtin(__builtin_amdgcn_tensor_load_to_lds)
#define USE_TDM 1
#else
#define USE_TDM 0
#endif

__device__ __forceinline__ void wait_tensor0() {
#if __has_builtin(__builtin_amdgcn_s_wait_tensorcnt)
  __builtin_amdgcn_s_wait_tensorcnt(0);
#else
  asm volatile("s_wait_tensorcnt 0x0" ::: "memory");
#endif
}

#if USE_TDM
// One-shot TDM copy of a contiguous 8KB (2048 x f32) block: global -> LDS.
// Descriptor packed per CDNA5 ISA ch.8 (D# groups): 1-D tile, data_size=4B.
__device__ __forceinline__ void tdm_copy_blk(const float* gsrc, float* ldst) {
  unsigned long long ga = (unsigned long long)(uintptr_t)gsrc;
  unsigned la = (unsigned)(uintptr_t)ldst;   // low 32 bits of generic = LDS byte addr
  v4u g0;
  g0[0] = 1u;                                 // count=1 (valid), user mode
  g0[1] = la;                                 // lds_addr
  g0[2] = (unsigned)ga;                       // global_addr[31:0]
  g0[3] = (unsigned)(ga >> 32) | (2u << 30);  // global_addr[56:32] | type=2
  v8i g1;
  g1[0] = (int)(2u << 16);                    // workgroup_mask=0, data_size=2 (4B)
  g1[1] = (int)(2048u << 16);                 // tensor_dim0[15:0] @ bits 63:48
  g1[2] = (int)(1u << 16);                    // tensor_dim0 hi=0; tensor_dim1=1
  g1[3] = (int)(2048u << 16);                 // tensor_dim1 hi=0; tile_dim0=2048
  g1[4] = 0;                                  // tile_dim1=0 (1-D), tile_dim2=0
  g1[5] = (int)2048u;                         // tensor_dim0_stride lo
  g1[6] = 0;                                  // stride hi, tensor_dim1_stride lo
  g1[7] = 0;
  v4i z4 = {0, 0, 0, 0};
#if __clang_major__ >= 23
  v8i z8 = {0, 0, 0, 0, 0, 0, 0, 0};
  __builtin_amdgcn_tensor_load_to_lds(g0, g1, z4, z4, z8, 0);
#else
  __builtin_amdgcn_tensor_load_to_lds(g0, g1, z4, z4, 0);
#endif
}
#endif

// Concatenate two aligned 8-half LDS chunks into one WMMA fragment.
__device__ __forceinline__ v16h make_frag(const _Float16* p0, const _Float16* p1) {
  v8h a = *(const v8h*)p0;
  v8h b = *(const v8h*)p1;
  v16h r;
#pragma unroll
  for (int i = 0; i < 8; ++i) { r[i] = a[i]; r[i + 8] = b[i]; }
  return r;
}

__global__ __launch_bounds__(64, 1)
void swa_fwd(const float* __restrict__ Q, const float* __restrict__ K,
             const float* __restrict__ V, float* __restrict__ O) {
  __shared__ _Float16 sQ[QBLK * KSTR];     // scaled Q block, f16
  __shared__ _Float16 sK[QBLK * KSTR];     // current K block, f16
  __shared__ _Float16 sVt[DIM * VSTR];     // current V block transposed, f16
  __shared__ _Float16 sP[2][16 * 32];      // per-wave P tile relay (C -> A layout)
#if USE_TDM
  __shared__ float sKraw[2][QBLK * DIM];   // TDM-landed raw f32, double buffered
  __shared__ float sVraw[2][QBLK * DIM];
#endif

  const int n    = blockIdx.x % NB;        // query block index
  const int bh   = blockIdx.x / NB;        // (batch, head)
  const size_t base = (size_t)bh * SEQ * DIM;
  const int tid  = threadIdx.x;
  const int wave = tid >> 5;               // 0,1 : query rows [wave*16, wave*16+16)
  const int lane = tid & 31;
  const int l16  = lane & 15;
  const int hi   = lane >> 4;

  int kb0 = n - (WIN - 1);
  if (kb0 < 0) kb0 = 0;

#if USE_TDM
  // Prologue: start DMA of the first key block while we stage Q.
  if (wave == 0) {
    tdm_copy_blk(K + base + (size_t)kb0 * QBLK * DIM, &sKraw[0][0]);
    tdm_copy_blk(V + base + (size_t)kb0 * QBLK * DIM, &sVraw[0][0]);
  }
#endif

  // ---- stage Q (pre-scaled by D^-1/2) into LDS as f16 ----
  {
    const float scale = 0.125f;            // 64^-0.5
    const float* q = Q + base + (size_t)n * QBLK * DIM;
#pragma unroll
    for (int i = tid; i < QBLK * DIM / 4; i += 64) {
      int r = i >> 4, c = (i & 15) * 4;
      f4 v = *(const f4*)(q + r * DIM + c);
      v4h h;
#pragma unroll
      for (int j = 0; j < 4; ++j) h[j] = (_Float16)(v[j] * scale);
      *(v4h*)&sQ[r * KSTR + c] = h;
    }
  }
  __syncthreads();

  // ---- A fragments: 16-bit A 16x32, contiguous k = 16*hi + s (B matches) ----
  v16h aQ[2];
  {
    int row = wave * 16 + l16;
#pragma unroll
    for (int kk = 0; kk < 2; ++kk) {
      int db = hi * 16 + kk * 32;
      aQ[kk] = make_frag(&sQ[row * KSTR + db], &sQ[row * KSTR + db + 8]);
    }
  }

  v8f acc[4];
#pragma unroll
  for (int j = 0; j < 4; ++j)
#pragma unroll
    for (int r = 0; r < 8; ++r) acc[j][r] = 0.0f;
  float mrow[8], lrow[8];
#pragma unroll
  for (int r = 0; r < 8; ++r) { mrow[r] = -1e30f; lrow[r] = 0.0f; }

  const float L2E = 1.4426950408889634f;
  int buf = 0;

  for (int kb = kb0; kb <= n; ++kb) {
#if USE_TDM
    if (wave == 0) wait_tensor0();         // raw[buf] DMA complete (issuing wave)
#endif
    __syncthreads();                       // publish raw[buf]; prior sK/sVt reads done

#if USE_TDM
    // Kick DMA for the NEXT key block; overlaps convert + WMMA below.
    if (wave == 0 && kb < n) {
      tdm_copy_blk(K + base + (size_t)(kb + 1) * QBLK * DIM, &sKraw[buf ^ 1][0]);
      tdm_copy_blk(V + base + (size_t)(kb + 1) * QBLK * DIM, &sVraw[buf ^ 1][0]);
    }
    // Convert raw f32 -> f16 (K row-major, V transposed) from LDS.
    {
      const float* kraw = &sKraw[buf][0];
      const float* vraw = &sVraw[buf][0];
#pragma unroll
      for (int i = tid; i < QBLK * DIM / 4; i += 64) {
        int r = i >> 4, c = (i & 15) * 4;
        f4 kv = *(const f4*)(kraw + r * DIM + c);
        v4h h;
#pragma unroll
        for (int j = 0; j < 4; ++j) h[j] = (_Float16)kv[j];
        *(v4h*)&sK[r * KSTR + c] = h;
        f4 vv = *(const f4*)(vraw + r * DIM + c);
#pragma unroll
        for (int j = 0; j < 4; ++j) sVt[(c + j) * VSTR + r] = (_Float16)vv[j];
      }
    }
#else
    // Fallback: direct global staging (no TDM builtin on this toolchain).
    {
      const float* kp = K + base + (size_t)kb * QBLK * DIM;
      const float* vp = V + base + (size_t)kb * QBLK * DIM;
#pragma unroll
      for (int i = tid; i < QBLK * DIM / 4; i += 64) {
        int r = i >> 4, c = (i & 15) * 4;
        f4 kv = *(const f4*)(kp + r * DIM + c);
        v4h h;
#pragma unroll
        for (int j = 0; j < 4; ++j) h[j] = (_Float16)kv[j];
        *(v4h*)&sK[r * KSTR + c] = h;
        f4 vv = *(const f4*)(vp + r * DIM + c);
#pragma unroll
        for (int j = 0; j < 4; ++j) sVt[(c + j) * VSTR + r] = (_Float16)vv[j];
      }
    }
#endif
    __syncthreads();

    // ---- scores: 16 q-rows x 32 keys, two 16x16 WMMA tiles ----
    v8f c0 = {}, c1 = {};
#pragma unroll
    for (int kk = 0; kk < 2; ++kk) {
      int db = hi * 16 + kk * 32;
      v16h b0 = make_frag(&sK[l16 * KSTR + db],        &sK[l16 * KSTR + db + 8]);
      v16h b1 = make_frag(&sK[(16 + l16) * KSTR + db], &sK[(16 + l16) * KSTR + db + 8]);
      c0 = __builtin_amdgcn_wmma_f32_16x16x32_f16(false, aQ[kk], false, b0, (short)0, c0, false, false);
      c1 = __builtin_amdgcn_wmma_f32_16x16x32_f16(false, aQ[kk], false, b1, (short)0, c1, false, false);
    }

    // ---- causal mask (diagonal block only) ----
    if (kb == n) {
      int qoff = wave * 16 + hi * 8;
#pragma unroll
      for (int r = 0; r < 8; ++r) {
        if (l16      > qoff + r) c0[r] = -1e30f;
        if (16 + l16 > qoff + r) c1[r] = -1e30f;
      }
    }

    // ---- online softmax (row stats in 16-lane halves; xor 1,2,4,8) ----
    _Float16* pp = &sP[wave][0];
#pragma unroll
    for (int r = 0; r < 8; ++r) {
      float s0 = c0[r], s1 = c1[r];
      float mx = fmaxf(s0, s1);
#pragma unroll
      for (int off = 8; off > 0; off >>= 1)
        mx = fmaxf(mx, __shfl_xor(mx, off, 32));
      float mn = fmaxf(mrow[r], mx);
      float al = exp2f((mrow[r] - mn) * L2E);
      float p0 = exp2f((s0 - mn) * L2E);
      float p1 = exp2f((s1 - mn) * L2E);
      float rs = p0 + p1;
#pragma unroll
      for (int off = 8; off > 0; off >>= 1)
        rs += __shfl_xor(rs, off, 32);
      lrow[r] = lrow[r] * al + rs;
      mrow[r] = mn;
#pragma unroll
      for (int j = 0; j < 4; ++j) acc[j][r] *= al;
      int m = r + hi * 8;
      pp[m * 32 + l16]      = (_Float16)p0;
      pp[m * 32 + 16 + l16] = (_Float16)p1;
    }
    __syncthreads();                       // P tile cross-lane visibility

    // ---- O += P * V ----
    v16h aP = make_frag(&pp[l16 * 32 + hi * 16], &pp[l16 * 32 + hi * 16 + 8]);
#pragma unroll
    for (int j = 0; j < 4; ++j) {
      int d = j * 16 + l16;
      v16h bV = make_frag(&sVt[d * VSTR + hi * 16], &sVt[d * VSTR + hi * 16 + 8]);
      acc[j] = __builtin_amdgcn_wmma_f32_16x16x32_f16(false, aP, false, bV, (short)0, acc[j], false, false);
    }
    buf ^= 1;
  }

  // ---- normalize and write ----
  float* out = O + base + (size_t)n * QBLK * DIM;
#pragma unroll
  for (int r = 0; r < 8; ++r) {
    float inv = 1.0f / lrow[r];
    int row = wave * 16 + hi * 8 + r;
#pragma unroll
    for (int j = 0; j < 4; ++j)
      out[row * DIM + j * 16 + l16] = acc[j][r] * inv;
  }
}

extern "C" void kernel_launch(void* const* d_in, const int* in_sizes, int n_in,
                              void* d_out, int out_size, void* d_ws, size_t ws_size,
                              hipStream_t stream) {
  const float* Q = (const float*)d_in[0];
  const float* K = (const float*)d_in[1];
  const float* V = (const float*)d_in[2];
  float* O = (float*)d_out;
  (void)in_sizes; (void)n_in; (void)out_size; (void)d_ws; (void)ws_size;
  dim3 grid(BSZ * NH * NB);   // 2048 workgroups, one per (b, h, query block)
  swa_fwd<<<grid, dim3(64), 0, stream>>>(Q, K, V, O);
}
